// DPLSacSMA_28509992911063
// MI455X (gfx1250) — matrix operand order repaired and checked by reading, not compiled, not run on hardware
//
#include <hip/hip_runtime.h>

typedef __attribute__((ext_vector_type(16))) _Float16 v16h;
typedef __attribute__((ext_vector_type(8)))  _Float16 v8h;
typedef __attribute__((ext_vector_type(8)))  float    v8f;
typedef __attribute__((ext_vector_type(4)))  float    v4f;

#define NSEQ 4000
#define TLEN 730
#define NB   200
#define HID  64
#define DDYN 8
#define DSTAT 16
#define ZDIM 256     // 4*HID
#define KC   96      // 8 (x) + 64 (h) padded to 3 chunks of 32
#define NCH  3
#define NPAR 30
#define UHLEN 15
#define NINC 4

__device__ __constant__ float PLO[NPAR] = {
  0.5f,0.7f,-2.0f,0.5f,0.05f,0.03f,0.0f,0.1f,0.02f,0.05f,0.0f,10.0f,5.0f,10.0f,10.0f,
  5.0f,0.1f,0.001f,0.01f,5.0f,1.0f,0.0f,0.0f,0.0f,0.0f,0.0f,0.0f,0.0f,1.0f,0.5f};
__device__ __constant__ float PHI[NPAR] = {
  2.0f,1.4f,2.0f,2.0f,0.49f,0.19f,1.0f,1.0f,0.3f,0.5f,0.3f,300.0f,150.0f,500.0f,1000.0f,
  400.0f,0.75f,0.05f,0.35f,350.0f,5.0f,0.8f,0.1f,0.4f,0.2f,0.5f,0.4f,1.0f,6.0f,5.0f};

__device__ __forceinline__ float fast_tanh(float x) {
#if __has_builtin(__builtin_amdgcn_tanhf)
  return __builtin_amdgcn_tanhf(x);
#else
  float e = __expf(2.0f * x);
  return (e - 1.0f) / (e + 1.0f);
#endif
}
__device__ __forceinline__ float sigm(float x) { return 1.0f / (1.0f + __expf(-x)); }

// Truncate generic pointer to the LDS byte offset (its low 32 bits).
__device__ __forceinline__ unsigned lds_addr32(const void* p) {
  return (unsigned)(unsigned long long)p;
}

// ---------------------------------------------------------------------------
// Pack [Wih | Whh] (f32) into CDNA5 WMMA B-fragment layout, f16.
// B 32x16 f16, contiguous K order used here (A reads match): lanes 0-15 hold column (lane), K = c*32 + 0..15;
//                     lanes 16-31 hold column (lane-16), K = c*32 + 16..31.
// flat = ((tile*NCH + c)*32 + lane)*16 + i
// ---------------------------------------------------------------------------
__global__ void prep_wpack(const float* __restrict__ Wih, const float* __restrict__ Whh,
                           _Float16* __restrict__ wpack) {
  int gid = blockIdx.x * blockDim.x + threadIdx.x;
  if (gid >= 16 * NCH * 32 * 16) return;
  int i    = gid & 15;
  int lane = (gid >> 4) & 31;
  int c    = (gid >> 9) % NCH;
  int tile = gid / (16 * 32 * NCH);
  int j = tile * 16 + (lane & 15);             // output column in z (0..255)
  int k = c * 32 + ((lane >> 4) ? 16 : 0) + i; // K in padded [x|h|0] (0..95)
  float v = 0.0f;
  if (k < DDYN)             v = Wih[j * DDYN + k];
  else if (k < DDYN + HID)  v = Whh[j * HID + (k - DDYN)];
  wpack[gid] = (_Float16)v;
}

// ---------------------------------------------------------------------------
// Static encoder: s = tanh(x_static @ Ws^T + bs)
// ---------------------------------------------------------------------------
__global__ void static_enc(const float* __restrict__ xs, const float* __restrict__ Ws,
                           const float* __restrict__ bs, float* __restrict__ senc) {
  int gid = blockIdx.x * blockDim.x + threadIdx.x;
  if (gid >= NSEQ * HID) return;
  int n = gid >> 6, j = gid & 63;
  float acc = bs[j];
#pragma unroll
  for (int d = 0; d < DSTAT; ++d) acc += xs[n * DSTAT + d] * Ws[j * DSTAT + d];
  senc[gid] = fast_tanh(acc);
}

// ---------------------------------------------------------------------------
// 32x32 tiled transpose of 32-bit words: (rows x cols) -> (cols x rows)
// ---------------------------------------------------------------------------
__global__ void transpose_k(const unsigned* __restrict__ src, unsigned* __restrict__ dst,
                            int rows, int cols) {
  __shared__ unsigned tile[32][33];
  int c0 = blockIdx.x * 32, r0 = blockIdx.y * 32;
  int x = threadIdx.x;
  for (int yy = threadIdx.y; yy < 32; yy += 8) {
    int r = r0 + yy, c = c0 + x;
    if (r < rows && c < cols) tile[yy][x] = src[(size_t)r * cols + c];
  }
  __syncthreads();
  for (int yy = threadIdx.y; yy < 32; yy += 8) {
    int r = c0 + yy, c = r0 + x;
    if (r < cols && c < rows) dst[(size_t)r * rows + c] = tile[x][yy];
  }
}

// ---------------------------------------------------------------------------
// LSTM: one block = 16 sequences, full T loop. 8 waves; wave w owns z-tiles
// 2w, 2w+1. B fragments live in VGPRs for all 730 steps. Forcing tile for
// step t+1 is staged LDS-ward by the async DMA engine (ASYNCcnt) while the
// WMMAs and gate math for step t execute.
// ---------------------------------------------------------------------------
__launch_bounds__(256)
__global__ void lstm_kernel(const float* __restrict__ xd, const _Float16* __restrict__ wpack,
                            const float* __restrict__ blstm, float* __restrict__ hfin) {
  __shared__ __attribute__((aligned(16))) _Float16 sA[16][KC]; // [x(8)|h(64)|pad(24)] f16
  __shared__ __attribute__((aligned(16))) float    sX[2][16][DDYN]; // async x staging (f32)
  __shared__ float sZ[16][ZDIM];   // z = A @ Wc^T (pre-activation)
  __shared__ float sC[16][HID];    // cell state
  __shared__ float sBias[ZDIM];

  int tid  = threadIdx.x;
  int lane = tid & 31, wv = tid >> 5;
  int n0 = blockIdx.x * 16;

  for (int p = tid; p < 16 * KC;  p += 256) (&sA[0][0])[p] = (_Float16)0.0f;
  for (int p = tid; p < 16 * HID; p += 256) (&sC[0][0])[p] = 0.0f;
  sBias[tid] = blstm[tid];

  // Preload B fragments (2 tiles x 3 K-chunks), 32B per lane each.
  v16h bf[2][NCH];
#pragma unroll
  for (int u = 0; u < 2; ++u) {
    int tile = wv * 2 + u;
#pragma unroll
    for (int c = 0; c < NCH; ++c)
      bf[u][c] = *(const v16h*)(wpack + ((size_t)((tile * NCH + c) * 32 + lane)) * 16);
  }

  // Async-stage x(t=0): wave 0, one b128 per lane (16 rows x 2 halves).
  if (tid < 32) {
    int m = tid >> 1, h4 = (tid & 1) * 4;
    unsigned ldsa = lds_addr32(&sX[0][m][h4]);
    const float* gp = &xd[((size_t)(n0 + m) * TLEN + 0) * DDYN + h4];
    asm volatile("global_load_async_to_lds_b128 %0, %1, off" :: "v"(ldsa), "v"(gp) : "memory");
  }
  __syncthreads();          // sA zero-fill complete everywhere
  asm volatile("s_wait_asynccnt 0" ::: "memory");
  if (tid < 16) {           // wave 0 converts its own completed copy
    v4f a0 = *(const v4f*)&sX[0][tid][0];
    v4f a1 = *(const v4f*)&sX[0][tid][4];
    v8h xo;
#pragma unroll
    for (int k = 0; k < 4; ++k) { xo[k] = (_Float16)a0[k]; xo[k + 4] = (_Float16)a1[k]; }
    *(v8h*)&sA[tid][0] = xo;
  }
  __syncthreads();

  for (int t = 0; t < TLEN; ++t) {
    // Kick off async copy of x(t+1) into the ping-pong buffer; the DMA
    // overlaps this step's WMMAs and gate math.
    if (t + 1 < TLEN && tid < 32) {
      int m = tid >> 1, h4 = (tid & 1) * 4;
      unsigned ldsa = lds_addr32(&sX[(t + 1) & 1][m][h4]);
      const float* gp = &xd[((size_t)(n0 + m) * TLEN + (t + 1)) * DDYN + h4];
      asm volatile("global_load_async_to_lds_b128 %0, %1, off" :: "v"(ldsa), "v"(gp) : "memory");
    }

    // A fragments from LDS, contiguous K order matching B pack:
    // lane<16:  row=lane,    K = c*32 + 0..15
    // lane>=16: row=lane-16, K = c*32 + 16..31
    v16h a[NCH];
    int row = lane & 15;
    int kb  = (lane >> 4) ? 16 : 0;
#pragma unroll
    for (int c = 0; c < NCH; ++c) {
      v8h lo = *(const v8h*)&sA[row][c * 32 + kb];
      v8h hi = *(const v8h*)&sA[row][c * 32 + kb + 8];
      v16h av;
#pragma unroll
      for (int i = 0; i < 8; ++i) { av[i] = lo[i]; av[i + 8] = hi[i]; }
      a[c] = av;
    }

#pragma unroll
    for (int u = 0; u < 2; ++u) {
      v8f acc = {};
#pragma unroll
      for (int c = 0; c < NCH; ++c)
        acc = __builtin_amdgcn_wmma_f32_16x16x32_f16(false, a[c], false, bf[u][c],
                                                     (short)0, acc, false, false);
      // C/D layout: lane<16 -> col=lane, rows 0..7; lane>=16 -> col=lane-16, rows 8..15
      int tile = wv * 2 + u;
      int col  = tile * 16 + (lane & 15);
      int m0   = (lane >> 4) * 8;
#pragma unroll
      for (int r = 0; r < 8; ++r) sZ[m0 + r][col] = acc[r];
    }
    __syncthreads();

    // Gate pass: 1024 (m,j) pairs over 256 threads.
#pragma unroll
    for (int q = 0; q < 4; ++q) {
      int p = tid * 4 + q;
      int m = p >> 6, j = p & 63;
      float zi = sZ[m][j]       + sBias[j];
      float zf = sZ[m][64 + j]  + sBias[64 + j];
      float zg = sZ[m][128 + j] + sBias[128 + j];
      float zo = sZ[m][192 + j] + sBias[192 + j];
      float cn = sigm(zf) * sC[m][j] + sigm(zi) * fast_tanh(zg);
      float hn = sigm(zo) * fast_tanh(cn);
      sC[m][j] = cn;
      sA[m][DDYN + j] = (_Float16)hn;
      if (t == TLEN - 1) hfin[(size_t)(n0 + m) * HID + j] = hn;
    }
    // Retire the async copy (wave 0 issued it; its ASYNCcnt gates completion)
    // and convert the staged f32 rows to the f16 A-tile for step t+1.
    if (t + 1 < TLEN) {
      asm volatile("s_wait_asynccnt 0" ::: "memory");
      if (tid < 16) {
        v4f a0 = *(const v4f*)&sX[(t + 1) & 1][tid][0];
        v4f a1 = *(const v4f*)&sX[(t + 1) & 1][tid][4];
        v8h xo;
#pragma unroll
        for (int k = 0; k < 4; ++k) { xo[k] = (_Float16)a0[k]; xo[k + 4] = (_Float16)a1[k]; }
        *(v8h*)&sA[tid][0] = xo;
      }
    }
    __syncthreads();
  }
}

// ---------------------------------------------------------------------------
// Head: raw = [h,s] @ Wout^T + bout -> sigmoid -> scaled params (stride 32)
// ---------------------------------------------------------------------------
__global__ void params_kernel(const float* __restrict__ hfin, const float* __restrict__ senc,
                              const float* __restrict__ Wout, const float* __restrict__ bout,
                              float* __restrict__ params) {
  int gid = blockIdx.x * blockDim.x + threadIdx.x;
  int n = gid >> 5, k = gid & 31;
  if (n >= NSEQ || k >= NPAR) return;
  float acc = bout[k];
#pragma unroll 8
  for (int j = 0; j < HID; ++j) acc += hfin[(size_t)n * HID + j] * Wout[k * 2 * HID + j];
#pragma unroll 8
  for (int j = 0; j < HID; ++j) acc += senc[(size_t)n * HID + j] * Wout[k * 2 * HID + HID + j];
  float g = sigm(acc);
  params[(size_t)n * 32 + k] = PLO[k] + g * (PHI[k] - PLO[k]);
}

__global__ void zero_k(float* __restrict__ p, int n) {
  int i = blockIdx.x * blockDim.x + threadIdx.x;
  if (i < n) p[i] = 0.0f;
}

// ---------------------------------------------------------------------------
// Fused Hamon PET + Snow-17 + SAC-SMA + Lohmann routing + basin segment-sum.
// One thread per unit; (T,N)-transposed forcing for coalesced streams.
// ---------------------------------------------------------------------------
__launch_bounds__(64)
__global__ void hydro_kernel(const float* __restrict__ prcpT, const float* __restrict__ tavgT,
                             const int* __restrict__ doyT, const float* __restrict__ params,
                             const float* __restrict__ elev, const float* __restrict__ latr,
                             const float* __restrict__ aw, const int* __restrict__ bidx,
                             float* __restrict__ qgauge, float* __restrict__ qbase,
                             float* __restrict__ qunit) {
  int n = blockIdx.x * blockDim.x + threadIdx.x;
  if (n >= NSEQ) return;
  const float* pp = params + (size_t)n * 32;
  float HAMON = pp[0],  SCF = pp[1],   PXTEMP = pp[2], MFMAX = pp[3], MFMIN = pp[4];
  float UADJ = pp[5],   MBASE = pp[6], TIPM = pp[7],   PLWHC = pp[8], NMF = pp[9];
  float DAYGM = pp[10], UZTWM = pp[11], UZFWM = pp[12], LZTWM = pp[13];
  float LZFPM = pp[14], LZFSM = pp[15], UZK = pp[16],  LZPK = pp[17], LZSK = pp[18];
  float ZPERC = pp[19], REXP = pp[20], PFREE = pp[21], PCTIM = pp[22], ADIMP = pp[23];
  float SIDE = pp[25],  UHN = pp[28],  UHTAU = pp[29];

  const float PI = 3.14159265358979f;
  float pa_fac = __expf(-elev[n] / 8434.0f);
  float tl = __tanf(latr[n]);

  float uh[UHLEN]; float wsum = 0.0f;
#pragma unroll
  for (int l = 0; l < UHLEN; ++l) {
    float tt = (float)(l + 1);
    float w = __expf((UHN - 1.0f) * __logf(tt) - tt / UHTAU);
    uh[l] = w; wsum += w;
  }
  float inv = 1.0f / wsum;
#pragma unroll
  for (int l = 0; l < UHLEN; ++l) uh[l] *= inv;

  float wi = 0.0f, wq = 0.0f, ati = 0.0f;
  float uztwc = 0.5f * UZTWM, uzfwc = 0.5f * UZFWM, lztwc = 0.5f * LZTWM;
  float lzfpc = 0.5f * LZFPM, lzfsc = 0.5f * LZFSM, adimc = 0.5f * (UZTWM + LZTWM);
  float percm = LZFPM * LZPK + LZFSM * LZSK;
  float lzmax = LZTWM + LZFPM + LZFSM;
  float hist[UHLEN - 1];
#pragma unroll
  for (int l = 0; l < UHLEN - 1; ++l) hist[l] = 0.0f;

  float w_ = aw[n]; int b = bidx[n];
  const float dt = 1.0f / NINC;

  for (int t = 0; t < TLEN; ++t) {
    float pr = prcpT[(size_t)t * NSEQ + n];
    float ta = tavgT[(size_t)t * NSEQ + n];
    float dy = (float)doyT[(size_t)t * NSEQ + n];
    if (t + 1 < TLEN) {
      __builtin_prefetch(&prcpT[(size_t)(t + 1) * NSEQ + n], 0, 1);
      __builtin_prefetch(&tavgT[(size_t)(t + 1) * NSEQ + n], 0, 1);
      __builtin_prefetch(&doyT[(size_t)(t + 1) * NSEQ + n], 0, 1);
    }
    // Hamon PET
    float decl = 0.4093f * __sinf(2.0f * PI / 365.0f * dy - 1.405f);
    float cosw = fminf(fmaxf(-tl * __tanf(decl), -0.9999f), 0.9999f);
    float daylen = acosf(cosw) * (24.0f / PI);
    float esat = 0.6108f * __expf(17.27f * ta / (ta + 237.3f));
    float pet = fmaxf(HAMON * 29.8f * daylen * esat / (ta + 273.2f), 0.0f);
    // Snow-17
    bool is_snow = (ta <= PXTEMP);
    float ps = is_snow ? pr * SCF : 0.0f;
    float prain = is_snow ? 0.0f : pr;
    wi += ps;
    float mf = 0.5f * (MFMAX + MFMIN) + 0.5f * (MFMAX - MFMIN) * __sinf(2.0f * PI * dy / 366.0f);
    ati += TIPM * (fminf(ta, 0.0f) - ati);
    float mros = UADJ * pa_fac * fmaxf(ta, 0.0f) * ((!is_snow && wi > 0.0f) ? prain : 0.0f) * 0.0125f;
    float melt = fminf(fmaxf(mf * fmaxf(ta - MBASE, 0.0f) + mros - NMF * fmaxf(-ati, 0.0f), 0.0f), wi);
    wi -= melt;
    float gm = fminf(DAYGM, wi); wi -= gm;
    float rop = (wi > 0.0f) ? prain : 0.0f;
    float rfree = prain - rop;
    wq += melt + rop;
    float outq = fmaxf(wq - PLWHC * wi, 0.0f); wq -= outq;
    float eff = outq + gm + rfree;
    // SAC-SMA
    float e1 = fminf(pet * uztwc / UZTWM, uztwc); uztwc -= e1;
    float red = pet - e1;
    float e2 = fminf(red, uzfwc); uzfwc -= e2; red -= e2;
    float e3 = fminf(red * lztwc / (UZTWM + LZTWM), lztwc); lztwc -= e3;
    float roimp = eff * PCTIM;
    float padimp = eff * ADIMP;
    adimc += padimp;
    float frac = fminf(fmaxf(adimc / (UZTWM + LZTWM), 0.0f), 1.0f);
    float adsur = padimp * frac * frac;
    adimc = fminf(adimc - adsur, UZTWM + LZTWM);
    float surf = roimp + adsur;
    float base = 0.0f;
    float pinc = eff * (1.0f - PCTIM - ADIMP) * dt;
#pragma unroll
    for (int it = 0; it < NINC; ++it) {
      float bfp = lzfpc * LZPK * dt;
      float bfs = lzfsc * LZSK * dt;
      lzfpc -= bfp; lzfsc -= bfs;
      base += (bfp + bfs) / (1.0f + SIDE);
      float qif = uzfwc * UZK * dt;
      uzfwc -= qif; surf += qif;
      float lzdef = LZTWM - lztwc + (LZFPM - lzfpc) + (LZFSM - lzfsc);
      float defr = fminf(fmaxf(lzdef / lzmax, 1e-6f), 1.0f);
      float perc = percm * (1.0f + ZPERC * __powf(defr, REXP)) * dt * uzfwc / UZFWM;
      perc = fminf(fminf(perc, uzfwc), fmaxf(lzdef, 0.0f));
      uzfwc -= perc;
      float pfree = perc * PFREE;
      lztwc += perc - pfree;
      float ex = fmaxf(lztwc - LZTWM, 0.0f);
      lztwc -= ex; pfree += ex;
      float dp = fmaxf(LZFPM - lzfpc, 0.0f);
      float ds = fmaxf(LZFSM - lzfsc, 0.0f);
      float fr = dp / fmaxf(dp + ds, 1e-6f);
      lzfpc += pfree * fr;
      lzfsc += pfree * (1.0f - fr);
      float exp_ = fmaxf(lzfpc - LZFPM, 0.0f);
      lzfpc -= exp_; lzfsc += exp_;
      float exs = fmaxf(lzfsc - LZFSM, 0.0f);
      lzfsc -= exs; surf += exs;
      uztwc += pinc;
      float exu = fmaxf(uztwc - UZTWM, 0.0f);
      uztwc -= exu; uzfwc += exu;
      float exf = fmaxf(uzfwc - UZFWM, 0.0f);
      uzfwc -= exf; surf += exf;
    }
    // Lohmann routing (register ring buffer) + outputs
    float q = base + uh[0] * surf;
#pragma unroll
    for (int l = 1; l < UHLEN; ++l) q += uh[l] * hist[l - 1];
#pragma unroll
    for (int l = UHLEN - 2; l >= 1; --l) hist[l] = hist[l - 1];
    hist[0] = surf;
    qunit[(size_t)n * TLEN + t] = q;
    unsafeAtomicAdd(&qgauge[(size_t)b * TLEN + t], q * w_);
    unsafeAtomicAdd(&qbase[(size_t)b * TLEN + t], base * w_);
  }
}

// ---------------------------------------------------------------------------
extern "C" void kernel_launch(void* const* d_in, const int* in_sizes, int n_in,
                              void* d_out, int out_size, void* d_ws, size_t ws_size,
                              hipStream_t stream) {
  const float* x_dyn    = (const float*)d_in[0];
  const float* x_static = (const float*)d_in[1];
  const float* prcp     = (const float*)d_in[2];
  const float* tavg     = (const float*)d_in[3];
  const int*   doy      = (const int*)d_in[4];
  const float* elev     = (const float*)d_in[5];
  const float* latr     = (const float*)d_in[6];
  const float* aw       = (const float*)d_in[7];
  const int*   bidx     = (const int*)d_in[8];
  // d_in[9] = n_basins (compile-time NB)
  const float* Wih   = (const float*)d_in[10];
  const float* Whh   = (const float*)d_in[11];
  const float* blstm = (const float*)d_in[12];
  const float* Ws    = (const float*)d_in[13];
  const float* bs    = (const float*)d_in[14];
  const float* Wout  = (const float*)d_in[15];
  const float* bout  = (const float*)d_in[16];

  float* out    = (float*)d_out;
  float* qgauge = out;
  float* qbase  = out + (size_t)NB * TLEN;
  float* qunit  = out + (size_t)2 * NB * TLEN;

  char* ws = (char*)d_ws;
  size_t off = 0;
  auto alloc = [&](size_t bytes) -> char* {
    char* p = ws + off;
    off = (off + bytes + 255) & ~(size_t)255;
    return p;
  };
  _Float16* wpack = (_Float16*)alloc((size_t)16 * NCH * 32 * 16 * sizeof(_Float16));
  float* senc   = (float*)alloc((size_t)NSEQ * HID * 4);
  float* hfin   = (float*)alloc((size_t)NSEQ * HID * 4);
  float* params = (float*)alloc((size_t)NSEQ * 32 * 4);
  float* prcpT  = (float*)alloc((size_t)NSEQ * TLEN * 4);
  float* tavgT  = (float*)alloc((size_t)NSEQ * TLEN * 4);
  int*   doyT   = (int*)alloc((size_t)NSEQ * TLEN * 4);

  prep_wpack<<<(16 * NCH * 32 * 16 + 255) / 256, 256, 0, stream>>>(Wih, Whh, wpack);
  static_enc<<<(NSEQ * HID + 255) / 256, 256, 0, stream>>>(x_static, Ws, bs, senc);

  dim3 tb(32, 8);
  dim3 tg((TLEN + 31) / 32, (NSEQ + 31) / 32);
  transpose_k<<<tg, tb, 0, stream>>>((const unsigned*)prcp, (unsigned*)prcpT, NSEQ, TLEN);
  transpose_k<<<tg, tb, 0, stream>>>((const unsigned*)tavg, (unsigned*)tavgT, NSEQ, TLEN);
  transpose_k<<<tg, tb, 0, stream>>>((const unsigned*)doy,  (unsigned*)doyT,  NSEQ, TLEN);

  lstm_kernel<<<NSEQ / 16, 256, 0, stream>>>(x_dyn, wpack, blstm, hfin);
  params_kernel<<<(NSEQ * 32 + 255) / 256, 256, 0, stream>>>(hfin, senc, Wout, bout, params);
  zero_k<<<(2 * NB * TLEN + 255) / 256, 256, 0, stream>>>(qgauge, 2 * NB * TLEN);
  hydro_kernel<<<(NSEQ + 63) / 64, 64, 0, stream>>>(prcpT, tavgT, doyT, params, elev, latr,
                                                    aw, bidx, qgauge, qbase, qunit);
}
